// EncodeProcessDecode_1649267441882
// MI455X (gfx1250) — matrix operand order repaired and checked by reading, not compiled, not run on hardware
//
#include <hip/hip_runtime.h>

#define NNODES 50000
#define NEDGES 400000
#define LATENT 64

typedef __bf16 bf16x16 __attribute__((ext_vector_type(16)));
typedef float  f32x8   __attribute__((ext_vector_type(8)));
typedef float  f32x4   __attribute__((ext_vector_type(4)));

// One MLP layer: out = [relu]( concat_segments(x0,x1,x2) @ W + b [+ resid] )
// Segments: KW0/KW1/KW2 actual widths (0 = unused), each padded to a multiple
// of 32 in WMMA-K space. NT = number of 16-wide N tiles, NOUT = actual cols.
// REQUIREMENT (guaranteed by the launcher for this model): nrows % 16 == 0.
// Each wave computes a 16-row x (NT*16)-col tile with v_wmma_f32_16x16x32_bf16.
template <int KW0, int KW1, int KW2, int NT, int NOUT, bool RELU, bool RESID>
__global__ __launch_bounds__(128) void mlp_layer(
    const float* __restrict__ x0, const int* __restrict__ idx0,
    const float* __restrict__ x1, const int* __restrict__ idx1,
    const float* __restrict__ x2, const int* __restrict__ idx2,
    const float* __restrict__ W,  const float* __restrict__ bias,
    const float* __restrict__ resid, float* __restrict__ out,
    int nrows)
{
    constexpr int C0 = (KW0 + 31) / 32, C1 = (KW1 + 31) / 32, C2 = (KW2 + 31) / 32;
    constexpr int P0 = 32 * C0, P1 = 32 * C1;
    constexpr int KPAD = 32 * (C0 + C1 + C2);
    constexpr int NCH = KPAD / 32;

    // Transposed bf16 weights in LDS: lW[n*KPAD + k]; B-fragment = 32 contiguous bytes.
    __shared__ __align__(32) __bf16 lW[NT * 16 * KPAD];
    __shared__ float lB[NT * 16];
    // Per-wave C-tile staging for the coalesced epilogue (only used when NT==4).
    __shared__ __align__(16) float lC[(NT == 4) ? 4 * 16 * 64 : 4];

    const int tid = threadIdx.x;
    for (int i = tid; i < NT * 16 * KPAD; i += 128) {
        int n = i / KPAD, kp = i - n * KPAD;
        float val = 0.0f;
        int arow = -1;
        if (kp < P0)            { if (kp < KW0) arow = kp; }
        else if (kp < P0 + P1)  { int kk = kp - P0;      if (kk < KW1) arow = KW0 + kk; }
        else                    { int kk = kp - P0 - P1; if (kk < KW2) arow = KW0 + KW1 + kk; }
        if (arow >= 0 && n < NOUT) val = W[(size_t)arow * NOUT + n];
        lW[i] = (__bf16)val;
    }
    for (int i = tid; i < NT * 16; i += 128) lB[i] = (i < NOUT) ? bias[i] : 0.0f;
    __syncthreads();

    const int lane = tid & 31;
    const int wave = tid >> 5;
    const int m    = lane & 15;      // A-fragment row within tile
    const int half = lane >> 4;      // K sub-group selector (ISA 16-bit A/B layouts)
    const int nl   = lane & 15;      // C/D column within N tile
    const int gw     = blockIdx.x * 4 + wave;
    const int nwaves = gridDim.x * 4;
    const int ntiles = nrows >> 4;   // full tiles only

    for (int tile = gw; tile < ntiles; tile += nwaves) {
        const int row0 = tile << 4;
        const int row  = row0 + m;
        int r0 = 0, r1 = 0, r2 = 0;
        if (KW0) r0 = idx0 ? idx0[row] : row;
        if (KW1) r1 = idx1 ? idx1[row] : row;
        if (KW2) r2 = idx2 ? idx2[row] : row;

        f32x8 acc[NT];
#pragma unroll
        for (int t = 0; t < NT; ++t) {
            float bv = lB[t * 16 + nl];
            acc[t] = (f32x8){bv, bv, bv, bv, bv, bv, bv, bv};
        }

#pragma unroll
        for (int c = 0; c < NCH; ++c) {
            int kloc = c * 32;
            const float* sp; int rr; int kw;
            if (kloc < P0)            { sp = x0; rr = r0; kw = KW0; }
            else if (kloc < P0 + P1)  { sp = x1; rr = r1; kw = KW1; kloc -= P0; }
            else                      { sp = x2; rr = r2; kw = KW2; kloc -= (P0 + P1); }
            const float* rowp = sp + (size_t)rr * kw;

            bf16x16 a;
            const bool full = (kloc + 32 <= kw);   // constant after unroll
            if (full) {
                // 16-bit A layout: K = kloc + half*16 + t (contiguous, matches B)
                f32x4 A0 = *(const f32x4*)(rowp + kloc + half * 16);
                f32x4 A1 = *(const f32x4*)(rowp + kloc + half * 16 + 4);
                f32x4 A2 = *(const f32x4*)(rowp + kloc + half * 16 + 8);
                f32x4 A3 = *(const f32x4*)(rowp + kloc + half * 16 + 12);
#pragma unroll
                for (int u = 0; u < 4; ++u) {
                    a[u]      = (__bf16)A0[u];
                    a[u + 4]  = (__bf16)A1[u];
                    a[u + 8]  = (__bf16)A2[u];
                    a[u + 12] = (__bf16)A3[u];
                }
            } else {
#pragma unroll
                for (int t = 0; t < 16; ++t) {
                    int kk = half * 16 + t;
                    int k  = kloc + kk;
                    float v = rowp[(k < kw) ? k : 0];
                    a[t] = (__bf16)((k < kw) ? v : 0.0f);
                }
            }

#pragma unroll
            for (int t = 0; t < NT; ++t) {
                // 16-bit B layout: lane column nl, K = c*32 + 16*half + (0..15) contiguous
                const __bf16* wp = &lW[(size_t)(t * 16 + nl) * KPAD + c * 32 + half * 16];
                bf16x16 b = *(const bf16x16*)wp;
                acc[t] = __builtin_amdgcn_wmma_f32_16x16x32_bf16(
                    false, a, false, b, (short)0, acc[t], false, false);
            }
        }

        if (NT == 4) {
            // Stage the 16x64 fp32 C tile in this wave's LDS region (C/D layout:
            // VGPR r holds row M = half*8 + r, column N = nl + 16*t), then re-read
            // row-major so residual loads and stores are coalesced b128s.
            float* cbuf = &lC[wave * 16 * 64];
#pragma unroll
            for (int t = 0; t < NT; ++t)
#pragma unroll
                for (int r = 0; r < 8; ++r)
                    cbuf[(half * 8 + r) * 64 + t * 16 + nl] = acc[t][r];
            // Same-wave LDS ops are processed in order; data-return waits are
            // inserted by the compiler (DScnt).
            const int lrow = lane >> 1;           // 0..15
            const int c0   = (lane & 1) * 32;     // 0 or 32
            const float* csrc = cbuf + lrow * 64 + c0;
            float* dst = out + (size_t)(row0 + lrow) * 64 + c0;
            const float* rsrc = RESID ? (resid + (size_t)(row0 + lrow) * 64 + c0) : nullptr;
#pragma unroll
            for (int q = 0; q < 8; ++q) {
                f32x4 v = *(const f32x4*)(csrc + q * 4);
                if (RESID) {
                    f32x4 rv = *(const f32x4*)(rsrc + q * 4);
                    v = v + rv;
                }
                if (RELU) {
#pragma unroll
                    for (int u = 0; u < 4; ++u) v[u] = fmaxf(v[u], 0.0f);
                }
                *(f32x4*)(dst + q * 4) = v;
            }
        } else {
            // Narrow-output path (decoder final layer, NOUT=3): scalar stores.
            int n = nl;
            if (n < NOUT) {
#pragma unroll
                for (int r = 0; r < 8; ++r) {
                    int orow = row0 + half * 8 + r;
                    float val = acc[0][r];
                    if (RESID) val += resid[(size_t)orow * NOUT + n];
                    if (RELU)  val = fmaxf(val, 0.0f);
                    out[(size_t)orow * NOUT + n] = val;
                }
            }
        }
    }
}

__global__ __launch_bounds__(256) void zero_f32(float* __restrict__ p, long long n)
{
    long long i = (long long)blockIdx.x * 256 + threadIdx.x;
    long long s = (long long)gridDim.x * 256;
    for (; i < n; i += s) p[i] = 0.0f;
}

__global__ __launch_bounds__(256) void scatter_add_edges(
    const float* __restrict__ e, const int* __restrict__ recv, float* __restrict__ agg)
{
    // One thread handles 4 consecutive latent columns of one edge.
    const long long total = (long long)NEDGES * (LATENT / 4);
    long long i = (long long)blockIdx.x * 256 + threadIdx.x;
    long long s = (long long)gridDim.x * 256;
    for (; i < total; i += s) {
        int eid = (int)(i >> 4);
        int c0  = (int)(i & 15) * 4;
        f32x4 v = *(const f32x4*)(e + (size_t)eid * LATENT + c0);
        float* base = agg + (size_t)recv[eid] * LATENT + c0;
        atomicAdd(base + 0, v[0]);
        atomicAdd(base + 1, v[1]);
        atomicAdd(base + 2, v[2]);
        atomicAdd(base + 3, v[3]);
    }
}

template <int KW0, int KW1, int KW2, int NT, int NOUT, bool RELU, bool RESID>
static void launch_layer(hipStream_t st,
    const float* x0, const int* i0, const float* x1, const int* i1,
    const float* x2, const int* i2,
    const float* W, const float* b, const float* resid, float* out, int nrows)
{
    int tiles = nrows / 16;
    int wgs = (tiles + 3) / 4;
    if (wgs > 1024) wgs = 1024;
    if (wgs < 1) wgs = 1;
    mlp_layer<KW0, KW1, KW2, NT, NOUT, RELU, RESID><<<wgs, 128, 0, st>>>(
        x0, i0, x1, i1, x2, i2, W, b, resid, out, nrows);
}

extern "C" void kernel_launch(void* const* d_in, const int* in_sizes, int n_in,
                              void* d_out, int out_size, void* d_ws, size_t ws_size,
                              hipStream_t stream)
{
    (void)in_sizes; (void)n_in; (void)out_size; (void)ws_size;

    const float* nodef = (const float*)d_in[0];
    const float* edgef = (const float*)d_in[1];
    const int*   send  = (const int*)d_in[2];
    const int*   recv  = send + NEDGES;
    auto Wp = [&](int i) { return (const float*)d_in[i]; };

    // Workspace: 3 edge buffers (E*64) + 3 node buffers + agg (N*64), fp32.
    float* ws = (float*)d_ws;
    const size_t eo = (size_t)NEDGES * LATENT, no = (size_t)NNODES * LATENT;
    float* e0 = ws;
    float* e1 = e0 + eo;
    float* e2 = e1 + eo;
    float* v0 = e2 + eo;
    float* v1 = v0 + no;
    float* v2 = v1 + no;
    float* agg = v2 + no;

    const float* Z = nullptr; const int* ZI = nullptr;

    // ---- Encoders ----
    launch_layer<16, 0, 0, 4, 64, true,  false>(stream, nodef, ZI, Z, ZI, Z, ZI, Wp(3), Wp(4), Z, v1, NNODES);
    launch_layer<64, 0, 0, 4, 64, true,  false>(stream, v1, ZI, Z, ZI, Z, ZI, Wp(5), Wp(6), Z, v2, NNODES);
    launch_layer<64, 0, 0, 4, 64, false, false>(stream, v2, ZI, Z, ZI, Z, ZI, Wp(7), Wp(8), Z, v0, NNODES);

    launch_layer<8,  0, 0, 4, 64, true,  false>(stream, edgef, ZI, Z, ZI, Z, ZI, Wp(9), Wp(10), Z, e1, NEDGES);
    launch_layer<64, 0, 0, 4, 64, true,  false>(stream, e1, ZI, Z, ZI, Z, ZI, Wp(11), Wp(12), Z, e2, NEDGES);
    launch_layer<64, 0, 0, 4, 64, false, false>(stream, e2, ZI, Z, ZI, Z, ZI, Wp(13), Wp(14), Z, e0, NEDGES);

    float* ecur = e0; float* eoth = e1;
    float* vcur = v0; float* voth = v1;

    // ---- Processor: 3 message-passing steps ----
    for (int s = 0; s < 3; ++s) {
        const int base = 15 + 12 * s;

        // edge MLP on fused concat(e, v[send], v[recv]) -> e_new (+ residual e)
        launch_layer<64, 64, 64, 4, 64, true,  false>(stream, ecur, ZI, vcur, send, vcur, recv,
                                                      Wp(base + 0), Wp(base + 1), Z, eoth, NEDGES);
        launch_layer<64, 0, 0, 4, 64, true,  false>(stream, eoth, ZI, Z, ZI, Z, ZI,
                                                    Wp(base + 2), Wp(base + 3), Z, e2, NEDGES);
        launch_layer<64, 0, 0, 4, 64, false, true >(stream, e2, ZI, Z, ZI, Z, ZI,
                                                    Wp(base + 4), Wp(base + 5), ecur, eoth, NEDGES);
        { float* t = ecur; ecur = eoth; eoth = t; }

        // segment_sum(e_new, recv) -> agg
        zero_f32<<<512, 256, 0, stream>>>(agg, (long long)no);
        scatter_add_edges<<<2048, 256, 0, stream>>>(ecur, recv, agg);

        // node MLP on fused concat(v, agg) -> v_new (+ residual v)
        launch_layer<64, 64, 0, 4, 64, true,  false>(stream, vcur, ZI, agg, ZI, Z, ZI,
                                                     Wp(base + 6), Wp(base + 7), Z, voth, NNODES);
        launch_layer<64, 0, 0, 4, 64, true,  false>(stream, voth, ZI, Z, ZI, Z, ZI,
                                                    Wp(base + 8), Wp(base + 9), Z, v2, NNODES);
        launch_layer<64, 0, 0, 4, 64, false, true >(stream, v2, ZI, Z, ZI, Z, ZI,
                                                    Wp(base + 10), Wp(base + 11), vcur, voth, NNODES);
        { float* t = vcur; vcur = voth; voth = t; }
    }

    // ---- Decoder ----
    launch_layer<64, 0, 0, 4, 64, true,  false>(stream, vcur, ZI, Z, ZI, Z, ZI, Wp(51), Wp(52), Z, v2, NNODES);
    launch_layer<64, 0, 0, 4, 64, true,  false>(stream, v2, ZI, Z, ZI, Z, ZI, Wp(53), Wp(54), Z, voth, NNODES);
    launch_layer<64, 0, 0, 1, 3,  false, false>(stream, voth, ZI, Z, ZI, Z, ZI, Wp(55), Wp(56), Z,
                                                (float*)d_out, NNODES);
}
